// SelfAttention_8246337208600
// MI455X (gfx1250) — compile-verified, matrix operand order checked by reading, not run on hardware
//
#include <hip/hip_runtime.h>
#include <hip/hip_bf16.h>

#define BATCH 64
#define SEQ   4096
#define UNITS 256

typedef __attribute__((ext_vector_type(16))) _Float16 v16h;
typedef __attribute__((ext_vector_type(8)))  float    v8f;

// ---- workspace layout (float offsets) ----
#define WS_WSROW_OFF   0                                   // 64*256 f32
#define WS_E_OFF       (BATCH*UNITS)                       // 64*4096 f32
#define WS_UW_OFF      (WS_E_OFF + BATCH*SEQ)              // 256*256 f16 = 32768 f32 slots
#define WS_PART_OFF    (WS_UW_OFF + (UNITS*UNITS)/2)       // 64*8*256 f32
// total = 442368 floats ~= 1.73 MB

// ---------------------------------------------------------------------------
// 1) w_s = s_prev @ W_w + W_b   (64x256, tiny)
// ---------------------------------------------------------------------------
__global__ void __launch_bounds__(UNITS) wsrow_kernel(
    const float* __restrict__ s_prev, const float* __restrict__ W_w,
    const float* __restrict__ W_b, float* __restrict__ out)
{
    __shared__ float sp[UNITS];
    const int b = blockIdx.x, t = threadIdx.x;
    sp[t] = s_prev[b * UNITS + t];
    __syncthreads();
    float acc = W_b[t];
    #pragma unroll 4
    for (int k = 0; k < UNITS; ++k)
        acc += sp[k] * W_w[k * UNITS + t];   // coalesced across t
    out[b * UNITS + t] = acc;
}

// ---------------------------------------------------------------------------
// 2) Pre-swizzle U_w (f32, KxN = 256x256) into WMMA B-fragment order (f16).
//    Fragment addressing: frag = (ntile*8 + kstep)*32 + lane, 16 f16 per lane.
//    16-bit operand 32-K layout (mirrors documented A layout):
//      lane<16 : K = kstep*32 + {idx<8 ? idx : idx+8}        (kloc 0..7,16..23)
//      lane>=16: K = kstep*32 + 8 + {idx<8 ? idx : idx+8}    (kloc 8..15,24..31)
//      N = ntile*16 + (lane & 15)
// ---------------------------------------------------------------------------
__global__ void __launch_bounds__(256) uw_convert_kernel(
    const float* __restrict__ U_w, _Float16* __restrict__ uw_sw)
{
    const int gid   = blockIdx.x * 256 + threadIdx.x;   // 0..65535
    const int idx   =  gid        & 15;
    const int lane  = (gid >> 4)  & 31;
    const int kstep = (gid >> 9)  & 7;
    const int ntile =  gid >> 12;
    const int n    = ntile * 16 + (lane & 15);
    const int kloc = (idx & 7) + ((idx & 8) ? 16 : 0) + ((lane & 16) ? 8 : 0);
    const int k    = kstep * 32 + kloc;
    uw_sw[gid] = (_Float16)U_w[k * UNITS + n];
}

// ---------------------------------------------------------------------------
// 3) Fused score kernel:  e[b,s] = V . tanh(w_s[b] + U(h[b,s])) + V_b
//    Block = 256 threads (8 waves). Block tile = 16 seq x 256 units.
//    Wave w accumulates unit tiles nt0=2w, nt1=2w+1 with
//    v_wmma_f32_16x16x32_f16 over K=256 (8 chained WMMAs per tile).
// ---------------------------------------------------------------------------
__global__ void __launch_bounds__(256) score_kernel(
    const float* __restrict__ hs, const float* __restrict__ wsrow,
    const _Float16* __restrict__ uw_sw, const float* __restrict__ U_b,
    const float* __restrict__ V_w, const float* __restrict__ V_b,
    float* __restrict__ e_out)
{
    __shared__ __align__(32) _Float16 hA[16 * UNITS];   // A fragments, 8 KB
    __shared__ float e_sm[16];

    const int b   = blockIdx.y;
    const int s0  = blockIdx.x * 16;
    const int tid = threadIdx.x;

    if (tid < 16) e_sm[tid] = V_b[0];

    // ---- stage 16x256 f32 hidden tile -> LDS as f16 in A-fragment order ----
    {
        const int m  = tid >> 4;              // seq row within tile
        const int kb = (tid & 15) << 4;       // 16 consecutive K per thread
        const float* src = hs + (((size_t)b * SEQ + s0 + m) * UNITS + kb);
        float v[16];
        #pragma unroll
        for (int j = 0; j < 16; ++j) v[j] = src[j];   // -> global_load_b128 x4
        #pragma unroll
        for (int j = 0; j < 16; ++j) {
            const int kk    = kb + j;
            const int kstep = kk >> 5;
            const int kloc  = kk & 31;
            const int lane  = m + ((kloc & 8) ? 16 : 0);
            const int idx   = (kloc & 7) + ((kloc & 16) ? 8 : 0);
            hA[(kstep * 32 + lane) * 16 + idx] = (_Float16)v[j];
        }
    }
    __syncthreads();

    const int wave = tid >> 5;
    const int lane = tid & 31;
    const int nt0  = wave * 2;
    const int nt1  = nt0 + 1;
    const v16h* uwv = (const v16h*)uw_sw;

    v8f c0 = {}; v8f c1 = {};
    #pragma unroll
    for (int kstep = 0; kstep < 8; ++kstep) {
        v16h a  = *(const v16h*)&hA[(kstep * 32 + lane) * 16];
        v16h b0 = uwv[(nt0 * 8 + kstep) * 32 + lane];
        v16h b1 = uwv[(nt1 * 8 + kstep) * 32 + lane];
        c0 = __builtin_amdgcn_wmma_f32_16x16x32_f16(false, a, false, b0,
                                                    (short)0, c0, false, false);
        c1 = __builtin_amdgcn_wmma_f32_16x16x32_f16(false, a, false, b1,
                                                    (short)0, c1, false, false);
    }

    // ---- epilogue: +w_s +U_b, tanh, dot with V, reduce over units ----
    // C/D layout: vgpr r, lane L -> M = r + 8*(L>=16), N = L & 15
    const int n  = lane & 15;
    const int mh = (lane >> 4) * 8;
    const float* wsb = wsrow + b * UNITS;
    const float add0 = wsb[nt0 * 16 + n] + U_b[nt0 * 16 + n];
    const float add1 = wsb[nt1 * 16 + n] + U_b[nt1 * 16 + n];
    const float vw0  = V_w[nt0 * 16 + n];
    const float vw1  = V_w[nt1 * 16 + n];
    #pragma unroll
    for (int r = 0; r < 8; ++r) {
        float contrib = tanhf(c0[r] + add0) * vw0 + tanhf(c1[r] + add1) * vw1;
        contrib += __shfl_xor(contrib, 8, 16);
        contrib += __shfl_xor(contrib, 4, 16);
        contrib += __shfl_xor(contrib, 2, 16);
        contrib += __shfl_xor(contrib, 1, 16);
        if (n == 0) atomicAdd(&e_sm[mh + r], contrib);
    }
    __syncthreads();
    if (tid < 16) e_out[(size_t)b * SEQ + s0 + tid] = e_sm[tid];
}

// ---------------------------------------------------------------------------
// 4) softmax over seq per batch; writes weights into d_out
// ---------------------------------------------------------------------------
__global__ void __launch_bounds__(256) softmax_kernel(
    const float* __restrict__ e, float* __restrict__ w_out)
{
    const int b = blockIdx.x, t = threadIdx.x;
    const float* er = e + (size_t)b * SEQ;
    __shared__ float red[256];

    float vals[16];
    float mx = -__builtin_inff();
    #pragma unroll
    for (int i = 0; i < 16; ++i) {
        vals[i] = er[t + i * 256];
        mx = fmaxf(mx, vals[i]);
    }
    red[t] = mx; __syncthreads();
    for (int s = 128; s > 0; s >>= 1) {
        if (t < s) red[t] = fmaxf(red[t], red[t + s]);
        __syncthreads();
    }
    mx = red[0]; __syncthreads();

    float sum = 0.f;
    #pragma unroll
    for (int i = 0; i < 16; ++i) {
        vals[i] = __expf(vals[i] - mx);
        sum += vals[i];
    }
    red[t] = sum; __syncthreads();
    for (int s = 128; s > 0; s >>= 1) {
        if (t < s) red[t] += red[t + s];
        __syncthreads();
    }
    const float inv = 1.0f / red[0];
    #pragma unroll
    for (int i = 0; i < 16; ++i)
        w_out[(size_t)b * SEQ + t + i * 256] = vals[i] * inv;
}

// ---------------------------------------------------------------------------
// 5) context partial sums: 8 seq-chunks x 64 batches, coalesced over units
// ---------------------------------------------------------------------------
__global__ void __launch_bounds__(256) ctx_partial_kernel(
    const float* __restrict__ hs, const float* __restrict__ w,
    float* __restrict__ partial)
{
    const int b = blockIdx.y, c = blockIdx.x, t = threadIdx.x;
    const float* wr = w  + (size_t)b * SEQ + c * 512;
    const float* hr = hs + ((size_t)b * SEQ + c * 512) * UNITS + t;
    float acc = 0.f;
    #pragma unroll 4
    for (int s = 0; s < 512; ++s)
        acc += wr[s] * hr[(size_t)s * UNITS];
    partial[((size_t)b * 8 + c) * UNITS + t] = acc;
}

// ---------------------------------------------------------------------------
// 6) deterministic final reduction into context output
// ---------------------------------------------------------------------------
__global__ void __launch_bounds__(256) ctx_reduce_kernel(
    const float* __restrict__ partial, float* __restrict__ ctx)
{
    const int b = blockIdx.x, t = threadIdx.x;
    float acc = 0.f;
    #pragma unroll
    for (int c = 0; c < 8; ++c)
        acc += partial[((size_t)b * 8 + c) * UNITS + t];
    ctx[b * UNITS + t] = acc;
}

// ---------------------------------------------------------------------------
extern "C" void kernel_launch(void* const* d_in, const int* in_sizes, int n_in,
                              void* d_out, int out_size, void* d_ws, size_t ws_size,
                              hipStream_t stream)
{
    const float* s_prev = (const float*)d_in[0];
    const float* hs     = (const float*)d_in[1];
    const float* W_w    = (const float*)d_in[2];
    const float* W_b    = (const float*)d_in[3];
    const float* U_w    = (const float*)d_in[4];
    const float* U_b    = (const float*)d_in[5];
    const float* V_w    = (const float*)d_in[6];
    const float* V_b    = (const float*)d_in[7];

    float* out_ctx = (float*)d_out;                 // (64,256)
    float* out_w   = out_ctx + BATCH * UNITS;       // (64,4096,1)

    float*    ws       = (float*)d_ws;
    float*    ws_wsrow = ws + WS_WSROW_OFF;
    float*    ws_e     = ws + WS_E_OFF;
    _Float16* ws_uw    = (_Float16*)(ws + WS_UW_OFF);
    float*    ws_part  = ws + WS_PART_OFF;

    wsrow_kernel      <<<dim3(BATCH),          dim3(UNITS), 0, stream>>>(s_prev, W_w, W_b, ws_wsrow);
    uw_convert_kernel <<<dim3(256),            dim3(256),   0, stream>>>(U_w, ws_uw);
    score_kernel      <<<dim3(SEQ / 16, BATCH),dim3(256),   0, stream>>>(hs, ws_wsrow, ws_uw,
                                                                         U_b, V_w, V_b, ws_e);
    softmax_kernel    <<<dim3(BATCH),          dim3(256),   0, stream>>>(ws_e, out_w);
    ctx_partial_kernel<<<dim3(8, BATCH),       dim3(256),   0, stream>>>(hs, out_w, ws_part);
    ctx_reduce_kernel <<<dim3(BATCH),          dim3(256),   0, stream>>>(ws_part, out_ctx);
}
